// PointNet2Encoder_14474039787793
// MI455X (gfx1250) — compile-verified, matrix operand order checked by reading, not run on hardware
//
#include <hip/hip_runtime.h>
#include <hip/hip_bf16.h>

#define BB 32
#define NN 8192
#define GG 512
#define KK 32
#define EPSB 1e-5f

typedef __attribute__((ext_vector_type(16))) _Float16 v16h;
typedef __attribute__((ext_vector_type(8)))  float    v8f;

// ---------------------------------------------------------------------------
// Prep: fold BatchNorm (inference) into weights/biases.
//   y = relu((Wx + b - m)*s + beta), s = gamma*rsqrt(var+eps)
//     = relu((W*s)x + ((b-m)*s + beta))
// Layer0 kept f32 (K=3, scalar). Layers 1/2 stored f16 for WMMA.
// ---------------------------------------------------------------------------
__global__ void prep_kernel(
    const float* w0, const float* b0, const float* g0, const float* be0, const float* m0, const float* v0,
    const float* w1, const float* b1, const float* g1, const float* be1, const float* m1, const float* v1,
    const float* w2, const float* b2, const float* g2, const float* be2, const float* m2, const float* v2,
    float* w0s, float* bias0, _Float16* w1h, float* bias1, _Float16* w2h, float* bias2)
{
    int tid = threadIdx.x; // 256 threads, 1 block
    for (int o = tid; o < 64; o += 256) {
        float s = g0[o] * rsqrtf(v0[o] + EPSB);
        bias0[o] = (b0[o] - m0[o]) * s + be0[o];
        w0s[o*3+0] = w0[o*3+0] * s;
        w0s[o*3+1] = w0[o*3+1] * s;
        w0s[o*3+2] = w0[o*3+2] * s;
    }
    for (int o = tid; o < 128; o += 256) {
        float s = g1[o] * rsqrtf(v1[o] + EPSB);
        bias1[o] = (b1[o] - m1[o]) * s + be1[o];
        for (int c = 0; c < 64; ++c) w1h[o*64+c] = (_Float16)(w1[o*64+c] * s);
    }
    for (int o = tid; o < 256; o += 256) {
        float s = g2[o] * rsqrtf(v2[o] + EPSB);
        bias2[o] = (b2[o] - m2[o]) * s + be2[o];
        for (int c = 0; c < 128; ++c) w2h[o*128+c] = (_Float16)(w2[o*128+c] * s);
    }
}

__global__ void zero_kernel(float* out, int n)
{
    int i = blockIdx.x * blockDim.x + threadIdx.x;
    if (i < n) out[i] = 0.0f;
}

// ---------------------------------------------------------------------------
// Farthest point sampling: one block (1024 thr = 32 waves) per batch.
// Each thread caches its 8 points AND its 8 mindist values in registers:
// the 511 dependent rounds are pure VALU + two 5-level wave32 xor-butterfly
// argmax reductions that carry the winner's coordinates, so no memory is
// touched on the critical path. 2 barriers per round.
// ---------------------------------------------------------------------------
__global__ __launch_bounds__(1024) void fps_kernel(const float* __restrict__ points,
                                                   float* __restrict__ centers)
{
    __shared__ float rv[32], rx[32], ry[32], rz[32];
    __shared__ int   ri[32];

    int b    = blockIdx.x;
    int tid  = threadIdx.x;
    int wave = tid >> 5;
    int lane = tid & 31;
    const float* P = points + (size_t)b * NN * 3;

    float px[8], py[8], pz[8], md[8];
#pragma unroll
    for (int t = 0; t < 8; ++t) {
        int j = tid + t * 1024;
        px[t] = P[j*3+0];
        py[t] = P[j*3+1];
        pz[t] = P[j*3+2];
        md[t] = 1e10f;
    }

    float cx = P[0], cy = P[1], cz = P[2];
    if (tid == 0) {
        centers[((size_t)b * GG) * 3 + 0] = cx;
        centers[((size_t)b * GG) * 3 + 1] = cy;
        centers[((size_t)b * GG) * 3 + 2] = cz;
    }

    for (int it = 1; it < GG; ++it) {
        // local argmax over 8 register-resident candidates (pure VALU)
        float bv = -1.0f, bx = 0.0f, by = 0.0f, bz = 0.0f;
        int bi = 0;
#pragma unroll
        for (int t = 0; t < 8; ++t) {
            float dx = px[t] - cx, dy = py[t] - cy, dz = pz[t] - cz;
            float d  = dx*dx + dy*dy + dz*dz;
            float m  = fminf(md[t], d);
            md[t] = m;
            if (m > bv) { bv = m; bi = tid + t*1024; bx = px[t]; by = py[t]; bz = pz[t]; }
        }
        // wave32 butterfly argmax carrying coordinates (result in all lanes)
#pragma unroll
        for (int off = 16; off > 0; off >>= 1) {
            float ov = __shfl_xor(bv, off, 32);
            int   oi = __shfl_xor(bi, off, 32);
            float ox = __shfl_xor(bx, off, 32);
            float oy = __shfl_xor(by, off, 32);
            float oz = __shfl_xor(bz, off, 32);
            if (ov > bv || (ov == bv && oi < bi)) { bv=ov; bi=oi; bx=ox; by=oy; bz=oz; }
        }
        if (lane == 0) { rv[wave]=bv; ri[wave]=bi; rx[wave]=bx; ry[wave]=by; rz[wave]=bz; }
        __syncthreads();
        // stage 2: every wave redundantly reduces the 32 wave winners;
        // butterfly leaves the global winner in every lane -> no broadcast.
        bv = rv[lane]; bi = ri[lane]; bx = rx[lane]; by = ry[lane]; bz = rz[lane];
#pragma unroll
        for (int off = 16; off > 0; off >>= 1) {
            float ov = __shfl_xor(bv, off, 32);
            int   oi = __shfl_xor(bi, off, 32);
            float ox = __shfl_xor(bx, off, 32);
            float oy = __shfl_xor(by, off, 32);
            float oz = __shfl_xor(bz, off, 32);
            if (ov > bv || (ov == bv && oi < bi)) { bv=ov; bi=oi; bx=ox; by=oy; bz=oz; }
        }
        if (tid == 0) {
            centers[((size_t)b * GG + it) * 3 + 0] = bx;
            centers[((size_t)b * GG + it) * 3 + 1] = by;
            centers[((size_t)b * GG + it) * 3 + 2] = bz;
        }
        cx = bx; cy = by; cz = bz;
        __syncthreads();   // protect rv/ri/rx/ry/rz reuse next round
    }
}

// ---------------------------------------------------------------------------
// kNN + group + recenter: one block (256 thr = 8 waves) per (b,g) center.
// Each thread keeps its 32 distances in REGISTERS (no 33 KB LDS array ->
// high occupancy). 32 argmin rounds: unrolled register scan + wave32
// butterfly + redundant 8-entry scan; the owning thread marks its slot
// (unrolled cndmask, no dynamic indexing) and writes the neighbor.
// ---------------------------------------------------------------------------
__global__ __launch_bounds__(256) void knn_kernel(const float* __restrict__ points,
                                                  const float* __restrict__ centers,
                                                  float* __restrict__ neigh)
{
    __shared__ float rv[8];
    __shared__ int   ri[8];

    int bg   = blockIdx.x;
    int b    = bg >> 9;          // / GG
    int tid  = threadIdx.x;
    int wave = tid >> 5;
    int lane = tid & 31;
    const float* P = points + (size_t)b * NN * 3;

    float cx = centers[(size_t)bg*3+0];
    float cy = centers[(size_t)bg*3+1];
    float cz = centers[(size_t)bg*3+2];

    float dr[32];
#pragma unroll
    for (int t = 0; t < 32; ++t) {
        int j = tid + t * 256;
        float dx = P[j*3+0] - cx;
        float dy = P[j*3+1] - cy;
        float dz = P[j*3+2] - cz;
        dr[t] = dx*dx + dy*dy + dz*dz;
    }

    for (int kk = 0; kk < KK; ++kk) {
        float bv = 3.0e38f; int bi = tid;
#pragma unroll
        for (int t = 0; t < 32; ++t) {
            int j = tid + t * 256;
            if (dr[t] < bv) { bv = dr[t]; bi = j; }
        }
#pragma unroll
        for (int off = 16; off > 0; off >>= 1) {
            float ov = __shfl_xor(bv, off, 32);
            int   oi = __shfl_xor(bi, off, 32);
            if (ov < bv || (ov == bv && oi < bi)) { bv = ov; bi = oi; }
        }
        if (lane == 0) { rv[wave] = bv; ri[wave] = bi; }
        __syncthreads();
        // all threads redundantly pick the global winner (no broadcast barrier)
        float wv = rv[0]; int wi = ri[0];
#pragma unroll
        for (int w = 1; w < 8; ++w) {
            float v = rv[w]; int i2 = ri[w];
            if (v < wv || (v == wv && i2 < wi)) { wv = v; wi = i2; }
        }
        int owner = wi & 255;
        if (tid == owner) {
            int slot = wi >> 8;
#pragma unroll
            for (int t = 0; t < 32; ++t)
                if (t == slot) dr[t] = 3.0e38f;   // unrolled -> cndmask, no spill
            size_t o = ((size_t)bg * KK + kk) * 3;
            neigh[o + 0] = P[wi*3+0] - cx;
            neigh[o + 1] = P[wi*3+1] - cy;
            neigh[o + 2] = P[wi*3+2] - cz;
        }
        __syncthreads();   // protect rv/ri reuse next round
    }
}

// ---------------------------------------------------------------------------
// Fused MLP (3->64->128->256) + BN + ReLU + max-pool via WMMA.
// One block (128 threads = 4 waves) per (b,g): M-tile = K=32 rows.
// A fragment: lane loads 16 contiguous f16 of row (lane&15), K-half by
// lane>>4 (contiguous k order, same for B). B fragment: weights stored [out][in] ==
// B^T row-major. B hoisted across mt; layer-2 A fragments hoisted across nt.
// ---------------------------------------------------------------------------
__global__ __launch_bounds__(128) void mlp_wmma_kernel(
    const float* __restrict__ neigh,
    const float* __restrict__ w0s, const float* __restrict__ bias0,
    const _Float16* __restrict__ w1h, const float* __restrict__ bias1,
    const _Float16* __restrict__ w2h, const float* __restrict__ bias2,
    float* __restrict__ out)
{
    __shared__ _Float16 h0[32 * 64];    // 4 KB
    __shared__ _Float16 h1[32 * 128];   // 8 KB

    int bg   = blockIdx.x;
    int b    = bg >> 9;
    int tid  = threadIdx.x;
    int wave = tid >> 5;
    int lane = tid & 31;
    int col  = lane & 15;        // row of A-tile / column of B/C/D-tile
    int half = lane >> 4;        // K-half select; also row-half of C/D

    // ---- layer 0: [32,3] -> [32,64], scalar FMAs ----
    const float* X = neigh + (size_t)bg * KK * 3;
    for (int e = tid; e < 32 * 64; e += 128) {
        int m = e >> 6, o = e & 63;
        float v = X[m*3+0] * w0s[o*3+0]
                + X[m*3+1] * w0s[o*3+1]
                + X[m*3+2] * w0s[o*3+2] + bias0[o];
        h0[m*64 + o] = (_Float16)fmaxf(v, 0.0f);
    }
    __syncthreads();

    // ---- layer 1: h1 = relu(h0 x W1 + bias1); B hoisted across mt ----
#pragma unroll
    for (int ntc = 0; ntc < 2; ++ntc) {
        int nt = wave * 2 + ntc;
        int n  = nt * 16 + col;
        v16h bw0 = *(const v16h*)&w1h[(size_t)n * 64 +  0 + half*16];
        v16h bw1 = *(const v16h*)&w1h[(size_t)n * 64 + 32 + half*16];
        float bb = bias1[n];
#pragma unroll
        for (int mt = 0; mt < 2; ++mt) {
            v16h a0 = *(const v16h*)&h0[(mt*16 + col) * 64 +  0 + half*16];
            v16h a1 = *(const v16h*)&h0[(mt*16 + col) * 64 + 32 + half*16];
            v8f acc = {};
            acc = __builtin_amdgcn_wmma_f32_16x16x32_f16(
                      false, a0, false, bw0, (short)0, acc, false, false);
            acc = __builtin_amdgcn_wmma_f32_16x16x32_f16(
                      false, a1, false, bw1, (short)0, acc, false, false);
#pragma unroll
            for (int j = 0; j < 8; ++j) {
                float v = fmaxf(acc[j] + bb, 0.0f);
                h1[(mt*16 + half*8 + j) * 128 + n] = (_Float16)v;
            }
        }
    }
    __syncthreads();

    // ---- layer 2: relu(h1 x W2 + bias2) + fused max-pool over 32 rows ----
    // A fragments hoisted into registers: reused by all 4 N-tiles of the wave.
    v16h a2[2][4];
#pragma unroll
    for (int mt = 0; mt < 2; ++mt)
#pragma unroll
        for (int q = 0; q < 4; ++q)
            a2[mt][q] = *(const v16h*)&h1[(mt*16 + col) * 128 + q*32 + half*16];

#pragma unroll
    for (int ntc = 0; ntc < 4; ++ntc) {
        int nt = wave * 4 + ntc;
        int n  = nt * 16 + col;
        v16h bw[4];
#pragma unroll
        for (int q = 0; q < 4; ++q)
            bw[q] = *(const v16h*)&w2h[(size_t)n * 128 + q*32 + half*16];
        float bb   = bias2[n];
        float cmax = 0.0f;               // relu >= 0, so 0 is the identity
#pragma unroll
        for (int mt = 0; mt < 2; ++mt) {
            v8f acc = {};
#pragma unroll
            for (int q = 0; q < 4; ++q)
                acc = __builtin_amdgcn_wmma_f32_16x16x32_f16(
                          false, a2[mt][q], false, bw[q], (short)0, acc, false, false);
#pragma unroll
            for (int j = 0; j < 8; ++j)
                cmax = fmaxf(cmax, fmaxf(acc[j] + bb, 0.0f));
        }
        // merge row-halves (lane <-> lane^16 share the same column n)
        cmax = fmaxf(cmax, __shfl_xor(cmax, 16, 32));
        if (half == 0) {
            // relu outputs are >= 0 -> IEEE bits are monotone as u32
            atomicMax((unsigned int*)&out[(size_t)b * 256 + n], __float_as_uint(cmax));
        }
    }
}

// ---------------------------------------------------------------------------
static inline size_t align_up(size_t x, size_t a) { return (x + a - 1) & ~(a - 1); }

extern "C" void kernel_launch(void* const* d_in, const int* in_sizes, int n_in,
                              void* d_out, int out_size, void* d_ws, size_t ws_size,
                              hipStream_t stream)
{
    const float* points = (const float*)d_in[0];
    const float* w0 = (const float*)d_in[1];
    const float* b0 = (const float*)d_in[2];
    const float* g0 = (const float*)d_in[3];
    const float* be0= (const float*)d_in[4];
    const float* m0 = (const float*)d_in[5];
    const float* v0 = (const float*)d_in[6];
    const float* w1 = (const float*)d_in[7];
    const float* b1 = (const float*)d_in[8];
    const float* g1 = (const float*)d_in[9];
    const float* be1= (const float*)d_in[10];
    const float* m1 = (const float*)d_in[11];
    const float* v1 = (const float*)d_in[12];
    const float* w2 = (const float*)d_in[13];
    const float* b2 = (const float*)d_in[14];
    const float* g2 = (const float*)d_in[15];
    const float* be2= (const float*)d_in[16];
    const float* m2 = (const float*)d_in[17];
    const float* v2 = (const float*)d_in[18];

    // workspace carve (~6.6 MB total)
    size_t off = 0;
    auto carve = [&](size_t bytes) -> void* {
        off = align_up(off, 256);
        void* p = (char*)d_ws + off;
        off += bytes;
        return p;
    };
    float*     centers = (float*)    carve((size_t)BB * GG * 3 * sizeof(float));
    float*     neigh   = (float*)    carve((size_t)BB * GG * KK * 3 * sizeof(float));
    float*     w0s     = (float*)    carve(64 * 3 * sizeof(float));
    float*     bias0   = (float*)    carve(64 * sizeof(float));
    _Float16*  w1h     = (_Float16*) carve(128 * 64 * sizeof(_Float16));
    float*     bias1   = (float*)    carve(128 * sizeof(float));
    _Float16*  w2h     = (_Float16*) carve(256 * 128 * sizeof(_Float16));
    float*     bias2   = (float*)    carve(256 * sizeof(float));

    float* out = (float*)d_out;

    // 1) fold BN into weights (f16) / biases
    prep_kernel<<<1, 256, 0, stream>>>(w0, b0, g0, be0, m0, v0,
                                       w1, b1, g1, be1, m1, v1,
                                       w2, b2, g2, be2, m2, v2,
                                       w0s, bias0, w1h, bias1, w2h, bias2);
    // 2) zero output (identity for relu-max)
    zero_kernel<<<(out_size + 255) / 256, 256, 0, stream>>>(out, out_size);
    // 3) farthest point sampling (register-resident, shuffle-reduced)
    fps_kernel<<<BB, 1024, 0, stream>>>(points, centers);
    // 4) kNN grouping + recenter (register-resident distances)
    knn_kernel<<<BB * GG, 256, 0, stream>>>(points, centers, neigh);
    // 5) fused WMMA MLP + BN + ReLU + max-pool
    mlp_wmma_kernel<<<BB * GG, 128, 0, stream>>>(neigh, w0s, bias0,
                                                 w1h, bias1, w2h, bias2, out);
}
